// MultiHeadAttention_27427661152385
// MI455X (gfx1250) — matrix operand order repaired and checked by reading, not compiled, not run on hardware
//
#include <hip/hip_runtime.h>
#include <hip/hip_bf16.h>

typedef _Float16 f16;
typedef __attribute__((ext_vector_type(16))) _Float16 v16h;
typedef __attribute__((ext_vector_type(8)))  float    v8f;

// ---------- helpers ----------
__device__ inline float wsum32(float v) {
#pragma unroll
  for (int o = 16; o > 0; o >>= 1) v += __shfl_xor(v, o, 32);
  return v;
}

// monotonic float<->uint mapping for atomicMax over signed floats
__device__ inline unsigned f2key(float f) {
  unsigned u = __float_as_uint(f);
  return (u & 0x80000000u) ? ~u : (u | 0x80000000u);
}
__device__ inline float key2f(unsigned k) {
  return (k & 0x80000000u) ? __uint_as_float(k & 0x7FFFFFFFu) : __uint_as_float(~k);
}

// A-matrix (16xK f16) pack per CDNA5 ISA: lane r=row (lane&15), g=lane>>4;
// v16h element i<8  -> K = k0 + 8g + i
// v16h element i>=8 -> K = k0 + 16 + 8g + (i-8)
__device__ inline v16h pack_a16(const f16* A, int ld, int r, int g, int k0) {
  v16h a;
  const f16* p0 = A + r * ld + k0 + 8 * g;
  const f16* p1 = p0 + 16;
#pragma unroll
  for (int i = 0; i < 8; ++i) a[i] = p0[i];
#pragma unroll
  for (int i = 0; i < 8; ++i) a[8 + i] = p1[i];
  return a;
}

// Load pre-swizzled B tile: 16 contiguous f16 per lane (two b128 loads).
__device__ inline v16h load_bp(const f16* __restrict__ Bp, int tile, int lane) {
  const f16* p = Bp + ((size_t)tile * 32 + lane) * 16;
  v16h b;
#pragma unroll
  for (int i = 0; i < 16; ++i) b[i] = p[i];
  return b;
}

// LayerNorm + SiLU over 128-wide rows of hb; lane owns row r, column half g.
__device__ inline void ln_silu_rows(float (*hb)[128], f16 (*as)[128],
                                    const float* gam, const float* bet, int r, int g) {
  float s = 0.f, ss = 0.f;
  for (int col = g * 64; col < g * 64 + 64; ++col) { float v = hb[r][col]; s += v; ss += v * v; }
  s  += __shfl_xor(s, 16, 32);
  ss += __shfl_xor(ss, 16, 32);
  float mu   = s * (1.f / 128.f);
  float var  = ss * (1.f / 128.f) - mu * mu;
  float rstd = rsqrtf(var + 1e-5f);
  for (int col = g * 64; col < g * 64 + 64; ++col) {
    float v  = (hb[r][col] - mu) * rstd * gam[col] + bet[col];
    float si = v * (1.f / (1.f + __expf(-v)));
    as[r][col] = (f16)si;
  }
}

// =====================================================================
// Kernel 0: pre-swizzle radial-MLP weights (f32, row-major [K][N]) into
// the CDNA5 WMMA B-operand lane layout as f16:
//   tile = kstep*NJ + ntile ; lane n=col (lane&15), group g=lane>>4 holds
//   K = 32*kstep + 8*g + i (i<8), 32*kstep + 16 + 8*g + (i-8) (i>=8)
// tiles: layer1 = 8 (KT=1,NJ=8), layer2 = 32 (KT=4,NJ=8), layer3 = 96 (KT=4,NJ=24)
// =====================================================================
__global__ __launch_bounds__(256)
void k_prep(const float* __restrict__ W1, const float* __restrict__ W2,
            const float* __restrict__ W3, f16* __restrict__ bp1,
            f16* __restrict__ bp2, f16* __restrict__ bp3) {
  int t = blockIdx.x * blockDim.x + threadIdx.x;
  if (t >= 136 * 32) return;
  int T = t >> 5, lane = t & 31;
  int g = lane >> 4, n = lane & 15;
  const float* W;
  f16* out;
  int ld, kt, j, tile;
  if (T < 8)       { W = W1; out = bp1; ld = 128; kt = 0;       j = T;            tile = T; }
  else if (T < 40) { int tt = T - 8;  W = W2; out = bp2; ld = 128; kt = tt >> 3;  j = tt & 7;  tile = tt; }
  else             { int tt = T - 40; W = W3; out = bp3; ld = 384; kt = tt / 24;  j = tt % 24; tile = tt; }
  int col = j * 16 + n;
  f16* p = out + ((size_t)tile * 32 + lane) * 16;
#pragma unroll
  for (int i = 0; i < 16; ++i)
    p[i] = (f16)W[(size_t)(kt * 32 + 8 * g + i + (i >= 8 ? 8 : 0)) * ld + col];
}

// =====================================================================
// Kernel 1: one wave (32 threads) per 16-edge tile.
//   Phase A: radial MLP via v_wmma_f32_16x16x32_f16 (32->128->128->384)
//   Phase B: per-edge rotate / so2 mix / so2 conv / eq-norm / logits
// =====================================================================
__global__ __launch_bounds__(32)
void k_edge(const float* __restrict__ ef,
            const float* __restrict__ x_src, const float* __restrict__ x_tar,
            const float* __restrict__ x_src0, const float* __restrict__ x_tar0,
            const float* __restrict__ RW, const int* __restrict__ edge,
            const f16* __restrict__ bp1, const float* __restrict__ b1,
            const float* __restrict__ g1, const float* __restrict__ be1,
            const f16* __restrict__ bp2, const float* __restrict__ b2,
            const float* __restrict__ g2, const float* __restrict__ be2,
            const f16* __restrict__ bp3, const float* __restrict__ b3,
            const float* __restrict__ qW, const float* __restrict__ qW0,
            const float* __restrict__ qb0,
            const float* __restrict__ kvW, const float* __restrict__ kvW0,
            const float* __restrict__ kvb0,
            const float* __restrict__ qng, const float* __restrict__ qng0,
            const float* __restrict__ kng, const float* __restrict__ kng0,
            float* __restrict__ glog, unsigned* __restrict__ mxk,
            f16* __restrict__ vst, int E) {
  __shared__ f16   Astage[16][128];
  __shared__ float hbuf[16][128];
  __shared__ f16   wall[16][384];
  __shared__ float sR[50], sXr[160], sQ[160], sQ0[32];
  __shared__ float sKV[320], sKV0[64], sK[160], sK0[32], sV[160], sV0[32];

  const int lane = threadIdx.x;
  const int r = lane & 15, g = lane >> 4;
  const int e0 = blockIdx.x * 16;

  // ---------------- Phase A: radial MLP ----------------
  // stage edge_feature tile (16 edges x 32) as f16
  {
    int e = e0 + r;
    for (int k = g * 16; k < g * 16 + 16; ++k)
      Astage[r][k] = (f16)(e < E ? ef[(size_t)e * 32 + k] : 0.f);
  }
  __syncthreads();

  // layer 1: K=32, N=128
  {
    v16h a = pack_a16(&Astage[0][0], 128, r, g, 0);
    for (int j = 0; j < 8; ++j) {
      int col = j * 16 + r;
      v16h bw = load_bp(bp1, j, lane);
      v8f c = {};
      c = __builtin_amdgcn_wmma_f32_16x16x32_f16(false, a, false, bw, (short)0, c, false, false);
#pragma unroll
      for (int v = 0; v < 8; ++v) hbuf[v + 8 * g][col] = c[v] + b1[col];
    }
  }
  __syncthreads();
  ln_silu_rows(hbuf, Astage, g1, be1, r, g);
  __syncthreads();

  // layer 2: K=128, N=128
#pragma unroll 1
  for (int j = 0; j < 8; ++j) {
    int col = j * 16 + r;
    v8f c = {};
    for (int kt = 0; kt < 4; ++kt) {
      v16h a  = pack_a16(&Astage[0][0], 128, r, g, kt * 32);
      v16h bw = load_bp(bp2, kt * 8 + j, lane);
      c = __builtin_amdgcn_wmma_f32_16x16x32_f16(false, a, false, bw, (short)0, c, false, false);
    }
#pragma unroll
    for (int v = 0; v < 8; ++v) hbuf[v + 8 * g][col] = c[v] + b2[col];
  }
  __syncthreads();
  ln_silu_rows(hbuf, Astage, g2, be2, r, g);
  __syncthreads();

  // layer 3: K=128, N=384 -> w_all (f16 in LDS)
#pragma unroll 1
  for (int j = 0; j < 24; ++j) {
    int col = j * 16 + r;
    v8f c = {};
    for (int kt = 0; kt < 4; ++kt) {
      v16h a  = pack_a16(&Astage[0][0], 128, r, g, kt * 32);
      v16h bw = load_bp(bp3, kt * 24 + j, lane);
      c = __builtin_amdgcn_wmma_f32_16x16x32_f16(false, a, false, bw, (short)0, c, false, false);
    }
#pragma unroll
    for (int v = 0; v < 8; ++v) wall[v + 8 * g][col] = (f16)(c[v] + b3[col]);
  }
  __syncthreads();

  // ---------------- Phase B: per-edge SO2 attention ----------------
  const int c2 = lane >> 1, l2 = lane & 1;   // (channel, l) pair owned by lane
#pragma unroll 1
  for (int el = 0; el < 16; ++el) {
    int e = e0 + el;
    if (e >= E) break;
    int sN = edge[e];
    int dN = edge[E + e];

    for (int i = lane; i < 50; i += 32) sR[i] = RW[(size_t)e * 50 + i];
    __syncthreads();

    // rotate target: xr[c,l,m] = sum_n R[l,m,n] * x_tar[dst][c,l,n]
    {
      const float* xt = x_tar + (size_t)dN * 160 + c2 * 10 + l2 * 5;
      float xv[5];
#pragma unroll
      for (int n = 0; n < 5; ++n) xv[n] = xt[n];
#pragma unroll
      for (int m = 0; m < 5; ++m) {
        float s = 0.f;
#pragma unroll
        for (int n = 0; n < 5; ++n) s += sR[l2 * 25 + m * 5 + n] * xv[n];
        sXr[c2 * 10 + l2 * 5 + m] = s;
      }
    }
    __syncthreads();

    // q mix: q[o,l,m] = sum_c qW[l,o,c]*xr[c,l,m];  q0 = x0 @ qW0^T + qb0
    {
#pragma unroll
      for (int m = 0; m < 5; ++m) {
        float s = 0.f;
        for (int c = 0; c < 16; ++c) s += qW[l2 * 256 + c2 * 16 + c] * sXr[c * 10 + l2 * 5 + m];
        sQ[c2 * 10 + l2 * 5 + m] = s;
      }
      float s0 = qb0[lane];
      const float* x0 = x_tar0 + (size_t)dN * 32;
      for (int i = 0; i < 32; ++i) s0 += qW0[lane * 32 + i] * x0[i];
      sQ0[lane] = s0;
    }
    __syncthreads();

    // eq_norm(q)
    {
      float ss = 0.f;
#pragma unroll
      for (int m = 0; m < 5; ++m) { float v = sQ[c2 * 10 + l2 * 5 + m]; ss += v * v; }
      ss = wsum32(ss);
      float gg = rsqrtf(ss * (1.f / 160.f) + 1e-5f) * qng[c2];
#pragma unroll
      for (int m = 0; m < 5; ++m) sQ[c2 * 10 + l2 * 5 + m] *= gg;
      float v0 = sQ0[lane];
      float mu = wsum32(v0) * (1.f / 32.f);
      float d  = v0 - mu;
      float var = wsum32(d * d) * (1.f / 32.f);
      sQ0[lane] = d * rsqrtf(var + 1e-5f) * qng0[lane];
    }
    __syncthreads();

    // rotate source (reuse sXr)
    {
      const float* xs = x_src + (size_t)sN * 160 + c2 * 10 + l2 * 5;
      float xv[5];
#pragma unroll
      for (int n = 0; n < 5; ++n) xv[n] = xs[n];
#pragma unroll
      for (int m = 0; m < 5; ++m) {
        float s = 0.f;
#pragma unroll
        for (int n = 0; n < 5; ++n) s += sR[l2 * 25 + m * 5 + n] * xv[n];
        sXr[c2 * 10 + l2 * 5 + m] = s;
      }
    }
    __syncthreads();

    // kv mix: 32 out channels x 2 l = 64 pairs
    for (int p = lane; p < 64; p += 32) {
      int o = p >> 1, l = p & 1;
#pragma unroll
      for (int m = 0; m < 5; ++m) {
        float s = 0.f;
        for (int c = 0; c < 16; ++c) s += kvW[l * 512 + o * 16 + c] * sXr[c * 10 + l * 5 + m];
        sKV[o * 10 + l * 5 + m] = s;
      }
    }
    {
      const float* x0 = x_src0 + (size_t)sN * 32;
      for (int j = lane; j < 64; j += 32) {
        float s = kvb0[j];
        for (int i = 0; i < 32; ++i) s += kvW0[j * 32 + i] * x0[i];
        sKV0[j] = s;
      }
    }
    __syncthreads();

    // so2_conv with per-edge weights w_all (wall[el])
    for (int p = lane; p < 64; p += 32) {
      int c = p >> 1, l = p & 1;
      float w1a = (float)wall[el][c * 8 + l * 4 + 0];
      float w1b = (float)wall[el][c * 8 + l * 4 + 1];
      float w2a = (float)wall[el][c * 8 + l * 4 + 2];
      float w2b = (float)wall[el][c * 8 + l * 4 + 3];
      const float* xx = &sKV[c * 10 + l * 5];
      float x0v = xx[0], x1v = xx[1], xm = xx[2], xp0 = xx[3], xp1 = xx[4];
      // xn = reversed(x[...,:2]) = (x1v, x0v)
      float yp0 = w1a * xp0 - w2a * x1v;
      float yp1 = w1b * xp1 - w2b * x0v;
      float yn0 = w2a * xp0 + w1a * x1v;
      float yn1 = w2b * xp1 + w1b * x0v;
      float ym  = (float)wall[el][320 + c * 2 + l] * xm;
      float* ob = (c < 16) ? &sK[c * 10 + l * 5] : &sV[(c - 16) * 10 + l * 5];
      ob[0] = yn1; ob[1] = yn0; ob[2] = ym; ob[3] = yp0; ob[4] = yp1;
    }
    sK0[lane] = (float)wall[el][256 + lane] * sKV0[lane];
    sV0[lane] = (float)wall[el][288 + lane] * sKV0[32 + lane];
    __syncthreads();

    // eq_norm(k)
    {
      float ss = 0.f;
#pragma unroll
      for (int m = 0; m < 5; ++m) { float v = sK[c2 * 10 + l2 * 5 + m]; ss += v * v; }
      ss = wsum32(ss);
      float gg = rsqrtf(ss * (1.f / 160.f) + 1e-5f) * kng[c2];
#pragma unroll
      for (int m = 0; m < 5; ++m) sK[c2 * 10 + l2 * 5 + m] *= gg;
      float v0 = sK0[lane];
      float mu = wsum32(v0) * (1.f / 32.f);
      float d  = v0 - mu;
      float var = wsum32(d * d) * (1.f / 32.f);
      sK0[lane] = d * rsqrtf(var + 1e-5f) * kng0[lane];
    }
    __syncthreads();

    // logits per head + segment max; store V (f16)
    if (lane < 8) {
      int h = lane;
      float s = 0.f;
#pragma unroll
      for (int ci = 0; ci < 2; ++ci) {
        int c = 2 * h + ci;
        for (int t = 0; t < 10; ++t) s += sQ[c * 10 + t] * sK[c * 10 + t];
      }
#pragma unroll
      for (int j = 0; j < 4; ++j) s += sQ0[4 * h + j] * sK0[4 * h + j];
      float lg = s * 0.70710678118654752f;   // / sqrt(d), d=2
      glog[(size_t)e * 8 + h] = lg;
      atomicMax(&mxk[dN * 8 + h], f2key(lg));
    }
    for (int i = lane; i < 160; i += 32) vst[(size_t)e * 192 + i] = (f16)sV[i];
    vst[(size_t)e * 192 + 160 + lane] = (f16)sV0[lane];
    __syncthreads();
  }
}

// =====================================================================
// Kernel 2: ex = exp(logit - segmax), accumulate denominators
// =====================================================================
__global__ __launch_bounds__(256)
void k_soft(const int* __restrict__ edge, const unsigned* __restrict__ mxk,
            float* __restrict__ glog, float* __restrict__ den, int E) {
  int t = blockIdx.x * blockDim.x + threadIdx.x;
  if (t >= E * 8) return;
  int e = t >> 3, h = t & 7;
  int dN = edge[E + e];
  float mx = key2f(mxk[dN * 8 + h]);
  float ex = __expf(glog[t] - mx);
  glog[t] = ex;
  atomicAdd(&den[dN * 8 + h], ex);
}

// =====================================================================
// Kernel 3: alpha * V, inverse rotation (R[l,n,m]), scatter-add to nodes
// one wave per edge
// =====================================================================
__global__ __launch_bounds__(256)
void k_scatter(const float* __restrict__ RW, const f16* __restrict__ vst,
               const float* __restrict__ glog, const float* __restrict__ den,
               const int* __restrict__ edge,
               float* __restrict__ att, float* __restrict__ att0, int E) {
  __shared__ float sR[8][50];
  int wid = threadIdx.x >> 5, lane = threadIdx.x & 31;
  int e = blockIdx.x * 8 + wid;
  bool act = e < E;
  int dN = 0;
  if (act) {
    dN = edge[E + e];
    for (int i = lane; i < 50; i += 32) sR[wid][i] = RW[(size_t)e * 50 + i];
  }
  __syncthreads();
  if (!act) return;

  int c = lane >> 1, l = lane & 1, h = c >> 1;
  float alpha = glog[(size_t)e * 8 + h] / den[dN * 8 + h];
  float vw[5];
#pragma unroll
  for (int n = 0; n < 5; ++n)
    vw[n] = alpha * (float)vst[(size_t)e * 192 + c * 10 + l * 5 + n];
#pragma unroll
  for (int m = 0; m < 5; ++m) {
    float s = 0.f;
#pragma unroll
    for (int n = 0; n < 5; ++n) s += sR[wid][l * 25 + n * 5 + m] * vw[n];
    atomicAdd(&att[(size_t)dN * 160 + c * 10 + l * 5 + m], s);
  }
  int h0 = lane >> 2;
  float a0 = glog[(size_t)e * 8 + h0] / den[dN * 8 + h0];
  atomicAdd(&att0[(size_t)dN * 32 + lane], a0 * (float)vst[(size_t)e * 192 + 160 + lane]);
}

// =====================================================================
// Kernel 4: final projections  y = einsum('loc,nclm->nolm', pW, att),
//           y0 = att0 @ pW0^T + pb0 ;  d_out = [y | y0]
// =====================================================================
__global__ __launch_bounds__(256)
void k_out(const float* __restrict__ att, const float* __restrict__ att0,
           const float* __restrict__ pW, const float* __restrict__ pW0,
           const float* __restrict__ pb0, float* __restrict__ out, int N) {
  int t = blockIdx.x * blockDim.x + threadIdx.x;
  if (t >= N * 32) return;
  int n = t >> 5, q = t & 31;
  if (q < 16) {
    int o = q;
#pragma unroll
    for (int l = 0; l < 2; ++l)
#pragma unroll
      for (int m = 0; m < 5; ++m) {
        float s = 0.f;
        for (int c = 0; c < 16; ++c)
          s += pW[l * 256 + o * 16 + c] * att[(size_t)n * 160 + c * 10 + l * 5 + m];
        out[(size_t)n * 160 + o * 10 + l * 5 + m] = s;
      }
  } else {
    int j = q - 16;
    for (int jj = j; jj < 32; jj += 16) {
      float s = pb0[jj];
      for (int i = 0; i < 32; ++i) s += pW0[jj * 32 + i] * att0[(size_t)n * 32 + i];
      out[(size_t)N * 160 + (size_t)n * 32 + jj] = s;
    }
  }
}

// =====================================================================
extern "C" void kernel_launch(void* const* d_in, const int* in_sizes, int n_in,
                              void* d_out, int out_size, void* d_ws, size_t ws_size,
                              hipStream_t stream) {
  const float* x_src  = (const float*)d_in[0];
  const float* x_tar  = (const float*)d_in[1];
  const float* x_src0 = (const float*)d_in[2];
  const float* x_tar0 = (const float*)d_in[3];
  const float* RW     = (const float*)d_in[4];
  const float* ef     = (const float*)d_in[5];
  const int*   edge   = (const int*)d_in[6];
  const float* W1  = (const float*)d_in[10];
  const float* b1  = (const float*)d_in[11];
  const float* g1  = (const float*)d_in[12];
  const float* be1 = (const float*)d_in[13];
  const float* W2  = (const float*)d_in[14];
  const float* b2  = (const float*)d_in[15];
  const float* g2  = (const float*)d_in[16];
  const float* be2 = (const float*)d_in[17];
  const float* W3  = (const float*)d_in[18];
  const float* b3  = (const float*)d_in[19];
  const float* qW   = (const float*)d_in[20];
  const float* qW0  = (const float*)d_in[21];
  const float* qb0  = (const float*)d_in[22];
  const float* kvW  = (const float*)d_in[23];
  const float* kvW0 = (const float*)d_in[24];
  const float* kvb0 = (const float*)d_in[25];
  const float* qng  = (const float*)d_in[26];
  const float* qng0 = (const float*)d_in[27];
  const float* kng  = (const float*)d_in[28];
  const float* kng0 = (const float*)d_in[29];
  const float* pW   = (const float*)d_in[30];
  const float* pW0  = (const float*)d_in[31];
  const float* pb0  = (const float*)d_in[32];

  const int N = in_sizes[0] / 160;   // x_src = (N,16,2,5)
  const int E = in_sizes[5] / 32;    // edge_feature = (E,32)

  // workspace carve-up
  char* w = (char*)d_ws;
  float* glog = (float*)w;            w += (size_t)E * 8 * sizeof(float);
  float* den  = (float*)w;            w += (size_t)N * 8 * sizeof(float);
  unsigned* mxk = (unsigned*)w;       w += (size_t)N * 8 * sizeof(unsigned);
  float* att  = (float*)w;            w += (size_t)N * 160 * sizeof(float);
  float* att0 = (float*)w;            w += (size_t)N * 32 * sizeof(float);
  f16* vst    = (f16*)w;              w += (size_t)E * 192 * sizeof(f16);
  f16* bp1    = (f16*)w;              w += (size_t)8  * 512 * sizeof(f16);
  f16* bp2    = (f16*)w;              w += (size_t)32 * 512 * sizeof(f16);
  f16* bp3    = (f16*)w;              w += (size_t)96 * 512 * sizeof(f16);

  // zero den + mxk + att + att0 (contiguous: N*208 dwords)
  hipMemsetAsync(den, 0, (size_t)N * 208 * sizeof(float), stream);

  // pre-swizzle weights into WMMA B layout (f16)
  k_prep<<<(136 * 32 + 255) / 256, 256, 0, stream>>>(W1, W2, W3, bp1, bp2, bp3);

  k_edge<<<(E + 15) / 16, 32, 0, stream>>>(
      ef, x_src, x_tar, x_src0, x_tar0, RW, edge,
      bp1, b1, g1, be1, bp2, b2, g2, be2, bp3, b3,
      qW, qW0, qb0, kvW, kvW0, kvb0, qng, qng0, kng, kng0,
      glog, mxk, vst, E);

  k_soft<<<(E * 8 + 255) / 256, 256, 0, stream>>>(edge, mxk, glog, den, E);

  k_scatter<<<(E + 7) / 8, 256, 0, stream>>>(RW, vst, glog, den, edge, att, att0, E);

  k_out<<<(N * 32 + 255) / 256, 256, 0, stream>>>(att, att0, pW, pW0, pb0,
                                                  (float*)d_out, N);
}
